// ReferenceAttention_89713276879676
// MI455X (gfx1250) — compile-verified, matrix operand order checked by reading, not run on hardware
//
#include <hip/hip_runtime.h>
#include <hip/hip_bf16.h>
#include <cstdint>
#include <cstddef>

typedef __attribute__((ext_vector_type(16))) __bf16 v16bf;
typedef __attribute__((ext_vector_type(8)))  __bf16 v8bf;
typedef __attribute__((ext_vector_type(4)))  __bf16 v4bf;
typedef __attribute__((ext_vector_type(8)))  float  v8f;
typedef __attribute__((ext_vector_type(4)))  float  v4f;

#define HD    1024
#define NHEAD 16
#define DKH   64
#define SEQ   2048
#define BATCH 4
#define N3    3072
#define MTOT  8192
#define BK    32
#define LDA   40   // LDS row stride (elems) for GEMM tiles (80B: bank-spread, 16B aligned)
#define LDF   72   // LDS row stride (elems) for flash tiles (144B)

// Async global->LDS DMA path (ASYNCcnt). Flip to 0 if the assembler rejects it.
#define USE_ASYNC 1

static __device__ __forceinline__ __bf16 to_bf16(float f) {
  union { float f; uint32_t u; } x; x.f = f;
  uint32_t r = x.u + 0x7fffu + ((x.u >> 16) & 1u);   // round-to-nearest-even
  uint16_t h = (uint16_t)(r >> 16);
  __bf16 out;
  __builtin_memcpy(&out, &h, 2);
  return out;
}

static __device__ __forceinline__ v16bf comb(v8bf lo, v8bf hi) {
  v16bf r;
#pragma unroll
  for (int i = 0; i < 8; ++i) { r[i] = lo[i]; r[i + 8] = hi[i]; }
  return r;
}

// Fragment load for WMMA A/B layout (ISA 7.12.2, 16-bit 16x32):
// lane<16 : row = lane,    k = sel*8 + (0..7) and sel*8 + (16..23), sel = 0
// lane>=16: same row, sel = 1
static __device__ __forceinline__ v16bf load_frag(const __bf16* p) {
  v8bf lo = *(const v8bf*)p;
  v8bf hi = *(const v8bf*)(p + 16);
  return comb(lo, hi);
}

static __device__ __forceinline__ void async_wait_all() {
#if USE_ASYNC
  asm volatile("s_wait_asynccnt 0" ::: "memory");
#endif
}

// Copy one 16B chunk global->LDS (async DMA if available, else VGPR round trip)
static __device__ __forceinline__ void copy16(__bf16* lds, const __bf16* gsrc) {
#if USE_ASYNC
  uint32_t loff = (uint32_t)(size_t)lds;               // low 32 bits = LDS offset
  uint64_t ga   = (uint64_t)(size_t)gsrc;
  asm volatile("global_load_async_to_lds_b128 %0, %1, off"
               :: "v"(loff), "v"(ga) : "memory");
#else
  *(v8bf*)lds = *(const v8bf*)gsrc;
#endif
}

// Stage a 128x32 bf16 tile (row stride HD in global) into LDS (stride LDA).
static __device__ __forceinline__ void stage_tile(__bf16* lds, const __bf16* gbase,
                                                  int k0, int tid) {
#pragma unroll
  for (int i = 0; i < 2; ++i) {
    int c   = tid + i * 256;          // 512 chunks of 8 elems
    int row = c >> 2;
    int col = (c & 3) * 8;
    copy16(lds + row * LDA + col, gbase + (size_t)row * HD + k0 + col);
  }
}

// ---------------------------------------------------------------------------
// Prep kernels: one-time fp32 -> bf16 conversions so GEMM tiles are pure copies
// ---------------------------------------------------------------------------
__global__ __launch_bounds__(256) void cvt_x_kernel(const float* __restrict__ in,
                                                    __bf16* __restrict__ out) {
  size_t i = (size_t)blockIdx.x * 256 + threadIdx.x;   // chunk of 4
  v4f v = *(const v4f*)(in + i * 4);
  v4bf h;
#pragma unroll
  for (int e = 0; e < 4; ++e) h[e] = to_bf16(v[e]);
  *(v4bf*)(out + i * 4) = h;
}

template <int N>   // W [1024 x N] fp32 -> Wt [N x 1024] bf16 (coalesced read)
__global__ __launch_bounds__(256) void wtr_kernel(const float* __restrict__ W,
                                                  __bf16* __restrict__ Wt) {
  size_t t = (size_t)blockIdx.x * 256 + threadIdx.x;
  int k = (int)(t / N);
  int n = (int)(t % N);
  Wt[(size_t)n * HD + k] = to_bf16(W[t]);
}

// ---------------------------------------------------------------------------
// GEMM: C[MTOT x N] = A[MTOT x 1024] * Bt[N x 1024]^T (+bias), bf16 WMMA, f32 acc
// Double-buffered LDS, async global->LDS staging, one barrier per K-step.
// EPI==0: scatter bias-added bf16 into head-major Q/K/V   EPI==1: fp32 C out
// ---------------------------------------------------------------------------
template <int EPI>
__global__ __launch_bounds__(256) void gemm_kernel(
    const __bf16* __restrict__ A, const __bf16* __restrict__ Bt,
    const float* __restrict__ bias,
    __bf16* __restrict__ Qo, __bf16* __restrict__ Ko, __bf16* __restrict__ Vo,
    float* __restrict__ Co)
{
  __shared__ __bf16 As[2][128 * LDA];     // [m][k]
  __shared__ __bf16 Bs[2][128 * LDA];     // [n][k]

  const int tid  = threadIdx.x;
  const int lane = tid & 31;
  const int wid  = tid >> 5;
  const int hm   = lane & 15;
  const int sel  = lane >> 4;
  const int bm   = blockIdx.y * 128;
  const int bn   = blockIdx.x * 128;
  const int wm   = (wid >> 2) * 64;
  const int wn   = (wid & 3) * 32;

  const __bf16* Abase = A  + (size_t)bm * HD;
  const __bf16* Bbase = Bt + (size_t)bn * HD;

  v8f acc[4][2] = {};

  stage_tile(&As[0][0], Abase, 0, tid);
  stage_tile(&Bs[0][0], Bbase, 0, tid);

  constexpr int T = HD / BK;              // 32 K-steps
  for (int t = 0; t < T; ++t) {
    async_wait_all();                     // my DMA for tile t done
    __syncthreads();                      // everyone's tile t visible; tile t-1 reads done
    if (t + 1 < T) {
      stage_tile(&As[(t + 1) & 1][0], Abase, (t + 1) * BK, tid);
      stage_tile(&Bs[(t + 1) & 1][0], Bbase, (t + 1) * BK, tid);
    }
    const __bf16* Asb = &As[t & 1][0];
    const __bf16* Bsb = &Bs[t & 1][0];

    v16bf afr[4], bfr[2];
#pragma unroll
    for (int i = 0; i < 4; ++i)
      afr[i] = load_frag(&Asb[(wm + i * 16 + hm) * LDA + sel * 8]);
#pragma unroll
    for (int j = 0; j < 2; ++j)
      bfr[j] = load_frag(&Bsb[(wn + j * 16 + hm) * LDA + sel * 8]);
#pragma unroll
    for (int i = 0; i < 4; ++i)
#pragma unroll
      for (int j = 0; j < 2; ++j)
        acc[i][j] = __builtin_amdgcn_wmma_f32_16x16x32_bf16(
            false, afr[i], false, bfr[j], (short)0, acc[i][j], false, false);
  }

  // ---- epilogue (C layout: VGPR e -> M = e + sel*8, N = hm) ----
#pragma unroll
  for (int i = 0; i < 4; ++i)
#pragma unroll
    for (int j = 0; j < 2; ++j)
#pragma unroll
      for (int e = 0; e < 8; ++e) {
        int m = bm + wm + i * 16 + e + sel * 8;
        int n = bn + wn + j * 16 + hm;
        float v = acc[i][j][e] + bias[n];
        if constexpr (EPI == 0) {
          int part = n >> 10;             // 0:Q 1:K 2:V
          int col  = n & 1023;
          int h    = col >> 6;
          int dk   = col & 63;
          int b    = m >> 11;             // SEQ = 2048
          int s    = m & 2047;
          __bf16* dst = (part == 0) ? Qo : (part == 1) ? Ko : Vo;
          dst[(((size_t)(b * NHEAD + h)) * SEQ + s) * DKH + dk] = to_bf16(v);
        } else {
          Co[(size_t)m * HD + n] = v;
        }
      }
}

// ---------------------------------------------------------------------------
// Flash attention: one workgroup (4 waves) = one 64-row Q block of one (b,h).
// K staged via async DMA; V transposed manually; online softmax; WMMA bf16.
// ---------------------------------------------------------------------------
__global__ __launch_bounds__(128) void flash_attn_kernel(
    const __bf16* __restrict__ Q, const __bf16* __restrict__ K,
    const __bf16* __restrict__ V, __bf16* __restrict__ O)
{
  __shared__ __bf16 Ks[64 * LDF];         // [kn][dk]
  __shared__ __bf16 Vs[64 * LDF];         // transposed: [dk][kn]
  __shared__ __bf16 Ps[4][16 * LDF];      // per-wave P re-layout buffer

  const int tid  = threadIdx.x;
  const int lane = tid & 31;
  const int wid  = tid >> 5;
  const int hm   = lane & 15;
  const int sel  = lane >> 4;
  const int qb   = blockIdx.x;
  const int bh   = blockIdx.y;            // b*NH + h
  const size_t base = (size_t)bh * SEQ * DKH;
  const __bf16* Qb = Q + base;
  const __bf16* Kb = K + base;
  const __bf16* Vb = V + base;
  const int q0 = qb * 64 + wid * 16;

  v16bf qfr[2];
#pragma unroll
  for (int s2 = 0; s2 < 2; ++s2) {
    const __bf16* p = Qb + (size_t)(q0 + hm) * DKH + s2 * 32 + sel * 8;
    qfr[s2] = comb(*(const v8bf*)p, *(const v8bf*)(p + 16));
  }

  v8f oacc[4] = {};
  float rm[8], rl[8];
#pragma unroll
  for (int e = 0; e < 8; ++e) { rm[e] = -__builtin_inff(); rl[e] = 0.f; }
  const float scale = 0.125f;             // 1/sqrt(64)

  for (int kb = 0; kb <= qb; ++kb) {
    // ---- stage K (async DMA) and V (transposed, manual) ----
#pragma unroll
    for (int i = 0; i < 4; ++i) {
      int c   = tid + i * 128;
      int row = c >> 3;
      int col = (c & 7) * 8;
      copy16(&Ks[row * LDF + col], Kb + (size_t)(kb * 64 + row) * DKH + col);
      v8bf vv = *(const v8bf*)(Vb + (size_t)(kb * 64 + row) * DKH + col);
#pragma unroll
      for (int e = 0; e < 8; ++e) Vs[(col + e) * LDF + row] = vv[e];
    }
    async_wait_all();
    __syncthreads();

    // ---- S = Q @ K^T (16 x 64 strip) ----
    v8f sacc[4] = {};
#pragma unroll
    for (int j = 0; j < 4; ++j)
#pragma unroll
      for (int s2 = 0; s2 < 2; ++s2) {
        v16bf kf = load_frag(&Ks[(j * 16 + hm) * LDF + s2 * 32 + sel * 8]);
        sacc[j] = __builtin_amdgcn_wmma_f32_16x16x32_bf16(
            false, qfr[s2], false, kf, (short)0, sacc[j], false, false);
      }

    // ---- scale + causal mask (diagonal block only) ----
#pragma unroll
    for (int j = 0; j < 4; ++j)
#pragma unroll
      for (int e = 0; e < 8; ++e) {
        float v = sacc[j][e] * scale;
        if (kb == qb) {
          int qi = q0 + e + sel * 8;
          int ki = kb * 64 + j * 16 + hm;
          if (ki > qi) v = -__builtin_inff();
        }
        sacc[j][e] = v;
      }

    // ---- online softmax; each row lives in 16 lanes of one half-wave ----
#pragma unroll
    for (int e = 0; e < 8; ++e) {
      float mx = sacc[0][e];
#pragma unroll
      for (int j = 1; j < 4; ++j) mx = fmaxf(mx, sacc[j][e]);
#pragma unroll
      for (int d = 1; d < 16; d <<= 1) mx = fmaxf(mx, __shfl_xor(mx, d, 32));
      float mnew  = fmaxf(rm[e], mx);
      float alpha = __expf(rm[e] - mnew);
      rm[e] = mnew;
      float sum = 0.f;
#pragma unroll
      for (int j = 0; j < 4; ++j) {
        float pv = __expf(sacc[j][e] - mnew);
        sacc[j][e] = pv;
        sum += pv;
      }
#pragma unroll
      for (int d = 1; d < 16; d <<= 1) sum += __shfl_xor(sum, d, 32);
      rl[e] = rl[e] * alpha + sum;
#pragma unroll
      for (int dd = 0; dd < 4; ++dd) oacc[dd][e] *= alpha;
    }

    // ---- re-layout P: C layout -> A layout via per-wave LDS buffer ----
#pragma unroll
    for (int j = 0; j < 4; ++j)
#pragma unroll
      for (int e = 0; e < 8; ++e)
        Ps[wid][(e + sel * 8) * LDF + j * 16 + hm] = to_bf16(sacc[j][e]);

    v16bf pfr[2];
#pragma unroll
    for (int s2 = 0; s2 < 2; ++s2)
      pfr[s2] = load_frag(&Ps[wid][hm * LDF + s2 * 32 + sel * 8]);

    // ---- O += P @ V ----
#pragma unroll
    for (int d = 0; d < 4; ++d)
#pragma unroll
      for (int s2 = 0; s2 < 2; ++s2) {
        v16bf vf = load_frag(&Vs[(d * 16 + hm) * LDF + s2 * 32 + sel * 8]);
        oacc[d] = __builtin_amdgcn_wmma_f32_16x16x32_bf16(
            false, pfr[s2], false, vf, (short)0, oacc[d], false, false);
      }
    __syncthreads();
  }

  // ---- normalize + merge heads: bf16 attn-out [B*S, H] ----
  const int b = bh >> 4;
  const int h = bh & 15;
#pragma unroll
  for (int d = 0; d < 4; ++d)
#pragma unroll
    for (int e = 0; e < 8; ++e) {
      int qi = q0 + e + sel * 8;
      int dk = d * 16 + hm;
      float v = oacc[d][e] / rl[e];
      O[((size_t)(b * SEQ + qi)) * HD + h * DKH + dk] = to_bf16(v);
    }
}

// ---------------------------------------------------------------------------
extern "C" void kernel_launch(void* const* d_in, const int* in_sizes, int n_in,
                              void* d_out, int out_size, void* d_ws, size_t ws_size,
                              hipStream_t stream) {
  const float* x      = (const float*)d_in[0];
  const float* W_attn = (const float*)d_in[1];
  const float* b_attn = (const float*)d_in[2];
  const float* W_proj = (const float*)d_in[3];
  const float* b_proj = (const float*)d_in[4];
  float* out = (float*)d_out;

  const size_t qkv_elems = (size_t)BATCH * NHEAD * SEQ * DKH;   // 8,388,608
  __bf16* Qw  = (__bf16*)d_ws;
  __bf16* Kw  = Qw + qkv_elems;
  __bf16* Vw  = Kw + qkv_elems;
  __bf16* XAO = Vw + qkv_elems;             // x_bf16 (GEMM1) then attn-out (flash/GEMM2)
  __bf16* WtA = XAO + (size_t)MTOT * HD;    // [3072 x 1024] bf16
  __bf16* WtP = WtA + (size_t)N3 * HD;      // [1024 x 1024] bf16

  // prep: one-time conversions / transposes
  cvt_x_kernel<<<(MTOT * HD / 4) / 256, 256, 0, stream>>>(x, XAO);
  wtr_kernel<N3><<<(HD * N3) / 256, 256, 0, stream>>>(W_attn, WtA);
  wtr_kernel<HD><<<(HD * HD) / 256, 256, 0, stream>>>(W_proj, WtP);

  // qkv = x @ W_attn + b_attn  -> head-major bf16 Q/K/V
  gemm_kernel<0><<<dim3(N3 / 128, MTOT / 128), 256, 0, stream>>>(
      XAO, WtA, b_attn, Qw, Kw, Vw, nullptr);
  // causal flash attention -> bf16 attn-out (overwrites x_bf16, which is dead)
  flash_attn_kernel<<<dim3(SEQ / 64, BATCH * NHEAD), 128, 0, stream>>>(Qw, Kw, Vw, XAO);
  // out = attn_out @ W_proj + b_proj (fp32)
  gemm_kernel<1><<<dim3(HD / 128, MTOT / 128), 256, 0, stream>>>(
      XAO, WtP, b_proj, nullptr, nullptr, nullptr, out);
}
